// MDRNN_basic_41695542509781
// MI455X (gfx1250) — matrix operand order repaired and checked by reading, not compiled, not run on hardware
//
#include <hip/hip_runtime.h>
#include <hip/hip_bf16.h>
#include <math.h>

typedef __attribute__((ext_vector_type(16))) __bf16 v16bf;
typedef __attribute__((ext_vector_type(8)))  float  v8f;

struct U128 { unsigned int x, y, z, w; };

union Frag {
  v16bf v;
  U128  q[2];
  unsigned short u[16];
};

__device__ __forceinline__ unsigned short f2bfu(float f) {
  unsigned u = __builtin_bit_cast(unsigned, f);
  u += 0x7FFFu + ((u >> 16) & 1u);
  return (unsigned short)(u >> 16);
}
// fast activations: one v_exp_f32 + one v_rcp_f32 each (saturate correctly)
__device__ __forceinline__ float sigm_fast(float x) {
  return __builtin_amdgcn_rcpf(1.0f + __expf(-x));
}
__device__ __forceinline__ float tanh_fast(float x) {
  return 1.0f - 2.0f * __builtin_amdgcn_rcpf(__expf(2.0f * x) + 1.0f);
}

#define BATCH 4096
#define HGRID 15
#define HID   20
#define GATES 100   // 5*HID
#define NPAD  112   // GATES padded to 7x16
#define KPAD  64    // (10 x)+(20 h_l)+(20 h_t) padded

// ---------------------------------------------------------------------------
// Kernel 1: pack [Wx | Whl | Wht] (f32) into bf16 WMMA-friendly K layout:
//   k 0..9   : Wx rows           (k10..15 zero)
//   k 16..35 : Whl rows 0..19    (k36..39 zero)
//   k 40..59 : Wht rows 0..19    (k60..63 zero)
// ---------------------------------------------------------------------------
__global__ void prepack_weights(const float* __restrict__ Wx,
                                const float* __restrict__ Whl,
                                const float* __restrict__ Wht,
                                unsigned short* __restrict__ wcat) {
  int t = blockIdx.x * blockDim.x + threadIdx.x;
  if (t >= 4 * KPAD * NPAD) return;
  int d = t / (KPAD * NPAD);
  int r = t % (KPAD * NPAD);
  int k = r / NPAD, n = r % NPAD;
  float v = 0.f;
  if (n < GATES) {
    if (k < 10)                 v = Wx [(d * 10  +  k      ) * GATES + n];
    else if (k >= 16 && k < 36) v = Whl[(d * HID + (k - 16)) * GATES + n];
    else if (k >= 40 && k < 60) v = Wht[(d * HID + (k - 40)) * GATES + n];
  }
  wcat[t] = f2bfu(v);
}

// ---------------------------------------------------------------------------
// Kernel 2: conv3x3(3->10) + bias + maxpool2x2 + tanh -> feat bf16
// feat layout: [15][15][4096][16]  (channels padded 10 -> 16 with zeros)
// ---------------------------------------------------------------------------
__global__ void conv_pool_tanh(const float* __restrict__ x,
                               const float* __restrict__ w,
                               const float* __restrict__ bias,
                               unsigned short* __restrict__ feat) {
  int idx = blockIdx.x * blockDim.x + threadIdx.x;
  if (idx >= BATCH * HGRID * HGRID) return;
  int b   = idx / (HGRID * HGRID);
  int rem = idx % (HGRID * HGRID);
  int oh = rem / HGRID, ow = rem % HGRID;

  float patch[3][4][4];
  const float* xb = x + (size_t)b * 3 * 32 * 32;
  #pragma unroll
  for (int ci = 0; ci < 3; ++ci)
    #pragma unroll
    for (int dy = 0; dy < 4; ++dy)
      #pragma unroll
      for (int dx = 0; dx < 4; ++dx)
        patch[ci][dy][dx] = xb[ci * 1024 + (2 * oh + dy) * 32 + (2 * ow + dx)];

  union { unsigned short u[16]; U128 q[2]; } out;
  for (int c = 0; c < 10; ++c) {
    float mx = -1e30f;
    #pragma unroll
    for (int dy = 0; dy < 2; ++dy)
      #pragma unroll
      for (int dx = 0; dx < 2; ++dx) {
        float s = bias[c];
        #pragma unroll
        for (int ci = 0; ci < 3; ++ci)
          #pragma unroll
          for (int ky = 0; ky < 3; ++ky)
            #pragma unroll
            for (int kx = 0; kx < 3; ++kx)
              s += patch[ci][dy + ky][dx + kx] * w[((c * 3 + ci) * 3 + ky) * 3 + kx];
        mx = fmaxf(mx, s);
      }
    out.u[c] = f2bfu(tanh_fast(mx));
  }
  for (int c = 10; c < 16; ++c) out.u[c] = 0;

  U128* dst = (U128*)(feat + ((size_t)rem * BATCH + b) * 16);
  dst[0] = out.q[0];
  dst[1] = out.q[1];
}

// ---------------------------------------------------------------------------
// Kernel 3: 2D-LSTM scan. grid = (256 batch-tiles, 4 directions), 256 threads.
// State rows shifted by +1: row 0 of h_state/c_state is a permanent zero
// column -> branch-free left-neighbor reads at sj==0.
//   cell sj reads  left = [sj],  top = [sj+1],  writes [sj+1]
// ---------------------------------------------------------------------------
__launch_bounds__(256)
__global__ void mdrnn_scan(const unsigned short* __restrict__ feat,
                           const unsigned short* __restrict__ wcat,
                           const float* __restrict__ bmd,
                           float* __restrict__ hcat) {
  __shared__ __align__(16) unsigned short h_state[HGRID + 1][16][32]; // bf16, hid padded to 32
  __shared__ float c_state[HGRID + 1][16][HID];
  __shared__ float g_lds[16][NPAD];

  const int btile = blockIdx.x;      // 0..255
  const int dir   = blockIdx.y;      // 0..3
  const int tid   = threadIdx.x;
  const int wv    = tid >> 5;        // wave 0..7
  const int lane  = tid & 31;
  const int lm    = lane & 15;
  const int hi    = (lane >> 4) & 1; // high half of wave

  for (int p = tid; p < (HGRID + 1) * 16 * 32; p += 256) ((unsigned short*)h_state)[p] = 0;
  for (int p = tid; p < (HGRID + 1) * 16 * HID; p += 256) ((float*)c_state)[p] = 0.f;

  // one-time per-wave B fragments (32x16 bf16 each) + bias column
  Frag b0, b1;
  float bias_n = 0.f;
  const int n = wv * 16 + lm;        // gate column owned by this lane
  if (wv < 7) {
    #pragma unroll
    for (int s = 0; s < 16; ++s) {
      int k = (s < 8) ? (hi ? 8 : 0) + s : 16 + (hi ? 8 : 0) + (s - 8); // dense 16-bit B layout: table K order per lane-half
      b0.u[s] = wcat[(dir * KPAD +      k) * NPAD + n];
      b1.u[s] = wcat[(dir * KPAD + 32 + k) * NPAD + n];
    }
    if (n < GATES) bias_n = bmd[dir * GATES + n];
  }

  // fixed (batch,hid) pairs for the elementwise phase (320 pairs, 256 threads)
  const int m0 = tid / HID, hd0 = tid % HID;                 // always valid
  const int m1 = (tid + 256) / HID, hd1 = (tid + 256) % HID; // valid iff tid < 64
  const bool two = (tid < 64);

  __syncthreads();

  const int flipH = (dir >> 1) & 1;  // flips=[(),(1,),(0,),(0,1)]
  const int flipW = dir & 1;

  for (int si = 0; si < HGRID; ++si) {
    for (int sj = 0; sj < HGRID; ++sj) {
      const int i = flipH ? (HGRID - 1 - si) : si;
      const int j = flipW ? (HGRID - 1 - sj) : sj;

      if (wv < 7) {
        // A fragments: 16-bit A 16x32 layout. 8-slot groups:
        //  chunk0: lo {k0..7  = x[0..7],     k16..23 = h_l[0..7]}
        //          hi {k8..15 = x[8..15],    k24..31 = h_l[8..15]}
        //  chunk1: lo {k32..39 = h_l[16..23], k48..55 = h_t[8..15]}
        //          hi {k40..47 = h_t[0..7],   k56..63 = h_t[16..23]}
        Frag a0, a1;
        const U128* xf = (const U128*)(feat +
            (((size_t)(i * HGRID + j) * BATCH) + btile * 16 + lm) * 16 + (hi ? 8 : 0));
        a0.q[0] = *xf;
        a0.q[1] = *(const U128*)&h_state[sj][lm][hi ? 8 : 0];
        a1.q[0] = hi ? *(const U128*)&h_state[sj + 1][lm][0]
                     : *(const U128*)&h_state[sj][lm][16];
        a1.q[1] = hi ? *(const U128*)&h_state[sj + 1][lm][16]
                     : *(const U128*)&h_state[sj + 1][lm][8];

        v8f acc = {};
        acc = __builtin_amdgcn_wmma_f32_16x16x32_bf16(false, a0.v, false, b0.v,
                                                      (short)0, acc, false, false);
        acc = __builtin_amdgcn_wmma_f32_16x16x32_bf16(false, a1.v, false, b1.v,
                                                      (short)0, acc, false, false);
        #pragma unroll
        for (int r = 0; r < 8; ++r)
          g_lds[r + (hi ? 8 : 0)][n] = acc[r] + bias_n;   // C/D layout: row=vgpr(+8), col=lane
      }
      __syncthreads();

      // gate nonlinearity + 2D-LSTM state update
      const bool corner = (si == HGRID - 1) && (sj == HGRID - 1);
      {
        float gi  = g_lds[m0][hd0];
        float gfl = g_lds[m0][HID + hd0];
        float gft = g_lds[m0][2 * HID + hd0];
        float go  = g_lds[m0][3 * HID + hd0];
        float gg  = g_lds[m0][4 * HID + hd0];
        float cl  = c_state[sj][m0][hd0];       // left (row 0 == zeros)
        float ct  = c_state[sj + 1][m0][hd0];   // top (prev grid row)
        float c   = sigm_fast(gfl) * cl + sigm_fast(gft) * ct + sigm_fast(gi) * tanh_fast(gg);
        float h   = sigm_fast(go) * tanh_fast(c);
        c_state[sj + 1][m0][hd0] = c;
        h_state[sj + 1][m0][hd0] = f2bfu(h);
        if (corner)
          hcat[((size_t)(btile * 16 + m0)) * (4 * HID) + dir * HID + hd0] = h;
      }
      if (two) {
        float gi  = g_lds[m1][hd1];
        float gfl = g_lds[m1][HID + hd1];
        float gft = g_lds[m1][2 * HID + hd1];
        float go  = g_lds[m1][3 * HID + hd1];
        float gg  = g_lds[m1][4 * HID + hd1];
        float cl  = c_state[sj][m1][hd1];
        float ct  = c_state[sj + 1][m1][hd1];
        float c   = sigm_fast(gfl) * cl + sigm_fast(gft) * ct + sigm_fast(gi) * tanh_fast(gg);
        float h   = sigm_fast(go) * tanh_fast(c);
        c_state[sj + 1][m1][hd1] = c;
        h_state[sj + 1][m1][hd1] = f2bfu(h);
        if (corner)
          hcat[((size_t)(btile * 16 + m1)) * (4 * HID) + dir * HID + hd1] = h;
      }
      __syncthreads();
    }
  }
}

// ---------------------------------------------------------------------------
// Kernel 4: final FC  [4096,80] @ [80,10] + bias  (trivial FLOPs -> VALU)
// ---------------------------------------------------------------------------
__global__ void fc_kernel(const float* __restrict__ hcat,
                          const float* __restrict__ fcw,
                          const float* __restrict__ fcb,
                          float* __restrict__ out) {
  int t = blockIdx.x * blockDim.x + threadIdx.x;
  if (t >= BATCH * 10) return;
  int b = t / 10, o = t % 10;
  float s = fcb[o];
  const float* hr = hcat + (size_t)b * 80;
  #pragma unroll 8
  for (int k = 0; k < 80; ++k) s += hr[k] * fcw[k * 10 + o];
  out[t] = s;
}

extern "C" void kernel_launch(void* const* d_in, const int* in_sizes, int n_in,
                              void* d_out, int out_size, void* d_ws, size_t ws_size,
                              hipStream_t stream) {
  (void)in_sizes; (void)n_in; (void)out_size; (void)ws_size;
  const float* x      = (const float*)d_in[0];
  const float* conv_w = (const float*)d_in[1];
  const float* conv_b = (const float*)d_in[2];
  const float* Wx     = (const float*)d_in[3];
  const float* Whl    = (const float*)d_in[4];
  const float* Wht    = (const float*)d_in[5];
  const float* b_md   = (const float*)d_in[6];
  const float* fc_w   = (const float*)d_in[7];
  const float* fc_b   = (const float*)d_in[8];
  float* out = (float*)d_out;

  // scratch layout (16B aligned slices)
  char* ws = (char*)d_ws;
  unsigned short* wcat = (unsigned short*)ws;                   // 4*64*112*2      = 57,344 B
  unsigned short* feat = (unsigned short*)(ws + 57344);         // 15*15*4096*16*2 = 29,491,200 B
  float*          hcat = (float*)(ws + 57344 + 29491200);       // 4096*80*4       = 1,310,720 B

  prepack_weights<<<(4 * KPAD * NPAD + 255) / 256, 256, 0, stream>>>(Wx, Whl, Wht, wcat);
  conv_pool_tanh<<<(BATCH * HGRID * HGRID + 255) / 256, 256, 0, stream>>>(x, conv_w, conv_b, feat);
  mdrnn_scan<<<dim3(BATCH / 16, 4), 256, 0, stream>>>(feat, wcat, b_md, hcat);
  fc_kernel<<<(BATCH * 10 + 255) / 256, 256, 0, stream>>>(hcat, fc_w, fc_b, out);
}
